// Attention_34849364640381
// MI455X (gfx1250) — matrix operand order repaired and checked by reading, not compiled, not run on hardware
//
#include <hip/hip_runtime.h>

typedef __bf16 bf16;
typedef __attribute__((ext_vector_type(16))) __bf16 v16bf;
typedef __attribute__((ext_vector_type(8)))  __bf16 v8bf;
typedef __attribute__((ext_vector_type(8)))  float  v8f;
typedef __attribute__((__vector_size__(4 * sizeof(int)))) int v4i;
typedef v4i __attribute__((address_space(1)))* gptr_v4i;   // global
typedef v4i __attribute__((address_space(3)))* lptr_v4i;   // LDS

#if defined(__gfx1250__) && __has_builtin(__builtin_amdgcn_global_load_async_to_lds_b128) && __has_builtin(__builtin_amdgcn_s_wait_asynccnt)
#define USE_ASYNC 1
#else
#define USE_ASYNC 0
#endif

// ---------------- WMMA fragment helpers (gfx1250 wave32 layouts) ----------------

static __device__ __forceinline__ v16bf pack16(v8bf lo, v8bf hi) {
  v16bf r;
#pragma unroll
  for (int i = 0; i < 8; ++i) { r[i] = lo[i]; r[8 + i] = hi[i]; }
  return r;
}

// A-fragment pattern: per lane, K = base+{0..7} and base+16+{0..7}
static __device__ __forceinline__ v16bf load_split16(const bf16* p) {
  return pack16(*(const v8bf*)p, *(const v8bf*)(p + 16));
}

// B-fragment pattern: per lane, K = base+{0..7} and base+16+{0..7}
static __device__ __forceinline__ v16bf load_contig16(const bf16* p) {
  return pack16(*(const v8bf*)p, *(const v8bf*)(p + 16));
}

static __device__ __forceinline__ v8f wmma_bf16(v16bf a, v16bf b, v8f c) {
  // (neg_a, A, neg_b, B, c_mod, C, reuse_a, reuse_b)
  return __builtin_amdgcn_wmma_f32_16x16x32_bf16(false, a, false, b, (short)0, c,
                                                 false, false);
}

// 16-byte global->LDS stage, async (ASYNCcnt) when available.
static __device__ __forceinline__ void stage16(const bf16* gsrc, bf16* ldst) {
#if USE_ASYNC
  __builtin_amdgcn_global_load_async_to_lds_b128(
      (gptr_v4i)(void*)gsrc, (lptr_v4i)(void*)ldst, 0, 0);
#else
  *(v8bf*)ldst = *(const v8bf*)gsrc;
#endif
}
static __device__ __forceinline__ void wait_async_le1() {
#if USE_ASYNC
  __builtin_amdgcn_s_wait_asynccnt(1);
#endif
}
static __device__ __forceinline__ void wait_async_0() {
#if USE_ASYNC
  __builtin_amdgcn_s_wait_asynccnt(0);
#endif
}

// ---------------- Block-cooperative 32x64-per-wave GEMM core ----------------
// Block = 128 threads (4 waves). All waves share A rows [m0, m0+32); wave w owns
// columns [n0, n0+64). A k-slice (32x32 bf16 = 2 KB) is staged into LDS by the
// whole block (async, double-buffered); B fragments stream from global (L2).
template <int K>
static __device__ __forceinline__ void gemm_core_32x64(
    const bf16* __restrict__ A, int lda, const bf16* __restrict__ Bt, int ldb,
    int m0, int n0, bf16* ldsA /* 2*32*32 bf16 */, v8f (&acc)[8]) {
  const int tid  = threadIdx.x;
  const int lane = tid & 31;
  const int l16  = lane & 15;
  const int half = lane >> 4;
  const int srow = tid >> 2;   // 0..31 : staged A row
  const int sseg = tid & 3;    // 16B segment within the 64B row slice

  const bf16* ga = A + (size_t)(m0 + srow) * lda + sseg * 8;
  const int lofs = srow * 32 + sseg * 8;
  const bf16* bp[4];
#pragma unroll
  for (int nb = 0; nb < 4; ++nb)
    bp[nb] = Bt + (size_t)(n0 + nb * 16 + l16) * ldb + half * 8;

  stage16(ga, ldsA + lofs);            // prologue: k-slice 0 -> buf 0
  constexpr int NS = K / 32;
  for (int i = 0; i < NS; ++i) {
    const int k0 = i * 32;
    if (i + 1 < NS) {
      stage16(ga + k0 + 32, ldsA + ((i + 1) & 1) * 1024 + lofs);
      wait_async_le1();                // current slice done, next in flight
    } else {
      wait_async_0();
    }
    __syncthreads();                   // LDS slice visible to all waves
    const bf16* la = ldsA + (i & 1) * 1024;
    const v16bf a0 = load_split16(la + l16 * 32 + half * 8);
    const v16bf a1 = load_split16(la + (l16 + 16) * 32 + half * 8);
    __builtin_prefetch((const void*)(bp[0] + k0 + 128), 0, 3);
#pragma unroll
    for (int nb = 0; nb < 4; ++nb) {
      const v16bf bfrag = load_contig16(bp[nb] + k0);
      acc[nb]     = wmma_bf16(a0, bfrag, acc[nb]);
      acc[4 + nb] = wmma_bf16(a1, bfrag, acc[4 + nb]);
    }
    __syncthreads();                   // all reads done before buffer reuse
  }
}

// ---------------- Conversion kernels ----------------

__global__ void cvt_kernel(const float* __restrict__ in, bf16* __restrict__ out, int n) {
  int i = blockIdx.x * 256 + threadIdx.x;
  if (i < n) out[i] = (bf16)in[i];
}

// W: [K][N] row-major f32  ->  Wt: [N][K] bf16
template <int K, int N>
__global__ void cvt_t_kernel(const float* __restrict__ W, bf16* __restrict__ Wt) {
  int i = blockIdx.x * 256 + threadIdx.x;
  if (i < K * N) {
    int n = i / K;
    int k = i - n * K;
    Wt[i] = (bf16)W[(size_t)k * N + n];
  }
}

// ---------------- QKV projection GEMM ----------------
// xb [8192][1024] bf16, wt [3072][1024] bf16 (transposed W_qkv).
// Q,K columns (c<2048) -> qk [8192][2048]; V (c>=2048) -> vt [b*16+h][64][2048].
__global__ __launch_bounds__(128) void qkv_gemm_kernel(
    const bf16* __restrict__ xb, const bf16* __restrict__ wt,
    const float* __restrict__ bias, bf16* __restrict__ qk, bf16* __restrict__ vtb) {
  __shared__ __align__(16) bf16 ldsA[2 * 32 * 32];
  const int lane = threadIdx.x & 31;
  const int wave = threadIdx.x >> 5;
  const int n0 = blockIdx.x * 256 + wave * 64;
  const int m0 = blockIdx.y * 32;
  v8f acc[8] = {};
  gemm_core_32x64<1024>(xb, 1024, wt, 1024, m0, n0, ldsA, acc);
  const int l16 = lane & 15;
  const int mo  = (lane >> 4) << 3;
#pragma unroll
  for (int mt = 0; mt < 2; ++mt)
#pragma unroll
    for (int nb = 0; nb < 4; ++nb)
#pragma unroll
      for (int r = 0; r < 8; ++r) {
        const int m = m0 + mt * 16 + mo + r;
        const int c = n0 + nb * 16 + l16;
        const float v = acc[mt * 4 + nb][r] + bias[c];
        if (c < 2048) {                     // Q or K region (uniform per tile)
          qk[(size_t)m * 2048 + c] = (bf16)v;
        } else {                            // V region: store transposed
          const int cc  = c - 2048;         // = h*64 + d
          const int bb  = m >> 11;
          const int tok = m & 2047;
          vtb[((size_t)bb * 1024 + cc) * 2048 + tok] = (bf16)v;
        }
      }
}

// ---------------- Flash attention (32 q-rows per wave) ----------------
// qk: [8192][2048] (cols 0..1023 = Q(h*64+d), 1024..2047 = K(h*64+d))
// vt: [b*16+h][64][2048]; aout: [8192][1024] bf16.
__global__ __launch_bounds__(128) void attn_kernel(
    const bf16* __restrict__ qk, const bf16* __restrict__ vt,
    bf16* __restrict__ aout) {
  __shared__ __align__(16) bf16 Plds[4][2][16 * 32];
  const int lane = threadIdx.x & 31;
  const int wave = threadIdx.x >> 5;
  const int bh = blockIdx.x;            // b*16 + h
  const int b = bh >> 4, h = bh & 15;
  const int q0 = (blockIdx.y * 4 + wave) * 32;
  const int half = lane >> 4;
  const int l16  = lane & 15;
  const float scale = 0.125f;           // 1/sqrt(64)

  // Q A-fragments: 2 row-sets x (K=64 as two 16x32 fragments)
  v16bf aq[2][2];
#pragma unroll
  for (int rs = 0; rs < 2; ++rs) {
    const bf16* qp =
        qk + (size_t)(b * 2048 + q0 + rs * 16 + l16) * 2048 + h * 64 + half * 8;
    aq[rs][0] = pack16(*(const v8bf*)qp,        *(const v8bf*)(qp + 16));
    aq[rs][1] = pack16(*(const v8bf*)(qp + 32), *(const v8bf*)(qp + 48));
  }

  float mrun[2][8], lrun[2][8];
#pragma unroll
  for (int rs = 0; rs < 2; ++rs)
#pragma unroll
    for (int r = 0; r < 8; ++r) { mrun[rs][r] = -3.0e38f; lrun[rs][r] = 0.0f; }
  v8f acc[2][4] = {};

  for (int kt = 0; kt < 2048; kt += 32) {
    // K fragments for this 32-key tile, shared by both row-sets
    const bf16* kp =
        qk + (size_t)(b * 2048 + kt + l16) * 2048 + 1024 + h * 64 + half * 8;
    const v16bf kb0 = load_contig16(kp);
    const v16bf kb1 = load_contig16(kp + 32);
    const bf16* kp2 = kp + (size_t)16 * 2048;
    const v16bf kc0 = load_contig16(kp2);
    const v16bf kc1 = load_contig16(kp2 + 32);

#pragma unroll
    for (int rs = 0; rs < 2; ++rs) {
      v8f s0 = {}, s1 = {};
      s0 = wmma_bf16(aq[rs][0], kb0, s0);
      s0 = wmma_bf16(aq[rs][1], kb1, s0);
      s1 = wmma_bf16(aq[rs][0], kc0, s1);
      s1 = wmma_bf16(aq[rs][1], kc1, s1);
      // online softmax; element r belongs to row (r + half*8) of this row-set
#pragma unroll
      for (int r = 0; r < 8; ++r) {
        float t = fmaxf(s0[r], s1[r]) * scale;
        t = fmaxf(t, __shfl_xor(t, 1));
        t = fmaxf(t, __shfl_xor(t, 2));
        t = fmaxf(t, __shfl_xor(t, 4));
        t = fmaxf(t, __shfl_xor(t, 8));
        const float mn  = fmaxf(mrun[rs][r], t);
        const float fac = __expf(mrun[rs][r] - mn);
        mrun[rs][r] = mn;
        const float p0 = __expf(s0[r] * scale - mn);
        const float p1 = __expf(s1[r] * scale - mn);
        float rsum = p0 + p1;
        rsum += __shfl_xor(rsum, 1);
        rsum += __shfl_xor(rsum, 2);
        rsum += __shfl_xor(rsum, 4);
        rsum += __shfl_xor(rsum, 8);
        lrun[rs][r] = lrun[rs][r] * fac + rsum;
#pragma unroll
        for (int nn = 0; nn < 4; ++nn) acc[rs][nn][r] = acc[rs][nn][r] * fac;
        const int prow = r + half * 8;
        Plds[wave][rs][prow * 32 + l16]      = (bf16)p0;
        Plds[wave][rs][prow * 32 + 16 + l16] = (bf16)p1;
      }
    }
    // re-fragment P (D-layout -> A-layout) through LDS
    v16bf ap[2];
#pragma unroll
    for (int rs = 0; rs < 2; ++rs) {
      const bf16* pp = &Plds[wave][rs][l16 * 32 + half * 8];
      ap[rs] = pack16(*(const v8bf*)pp, *(const v8bf*)(pp + 16));
    }
    // O += P @ V ; V fragments (contiguous in keys) reused by both row-sets
#pragma unroll
    for (int nn = 0; nn < 4; ++nn) {
      const bf16* vp =
          vt + ((size_t)bh * 64 + nn * 16 + l16) * 2048 + kt + half * 8;
      const v16bf bv = load_contig16(vp);
      acc[0][nn] = wmma_bf16(ap[0], bv, acc[0][nn]);
      acc[1][nn] = wmma_bf16(ap[1], bv, acc[1][nn]);
    }
  }
  // epilogue: divide by l, merge heads into [token][1024]
#pragma unroll
  for (int rs = 0; rs < 2; ++rs)
#pragma unroll
    for (int r = 0; r < 8; ++r) {
      const float inv = 1.0f / lrun[rs][r];
      const size_t orow =
          (size_t)(b * 2048 + q0 + rs * 16 + r + half * 8) * 1024 + h * 64;
#pragma unroll
      for (int nn = 0; nn < 4; ++nn)
        aout[orow + nn * 16 + l16] = (bf16)(acc[rs][nn][r] * inv);
    }
}

// ---------------- Output projection GEMM ----------------
__global__ __launch_bounds__(128) void proj_gemm_kernel(
    const bf16* __restrict__ ain, const bf16* __restrict__ wt,
    const float* __restrict__ bias, float* __restrict__ out) {
  __shared__ __align__(16) bf16 ldsA[2 * 32 * 32];
  const int lane = threadIdx.x & 31;
  const int wave = threadIdx.x >> 5;
  const int n0 = blockIdx.x * 256 + wave * 64;
  const int m0 = blockIdx.y * 32;
  v8f acc[8] = {};
  gemm_core_32x64<1024>(ain, 1024, wt, 1024, m0, n0, ldsA, acc);
  const int l16 = lane & 15;
  const int mo  = (lane >> 4) << 3;
#pragma unroll
  for (int mt = 0; mt < 2; ++mt)
#pragma unroll
    for (int nb = 0; nb < 4; ++nb)
#pragma unroll
      for (int r = 0; r < 8; ++r) {
        const int m = m0 + mt * 16 + mo + r;
        const int c = n0 + nb * 16 + l16;
        out[(size_t)m * 1024 + c] = acc[mt * 4 + nb][r] + bias[c];
      }
}

// ---------------- Host launcher ----------------
extern "C" void kernel_launch(void* const* d_in, const int* in_sizes, int n_in,
                              void* d_out, int out_size, void* d_ws, size_t ws_size,
                              hipStream_t stream) {
  const float* x      = (const float*)d_in[0];
  const float* W_qkv  = (const float*)d_in[1];
  const float* b_qkv  = (const float*)d_in[2];
  const float* W_proj = (const float*)d_in[3];
  const float* b_proj = (const float*)d_in[4];
  float* out = (float*)d_out;

  // workspace partition (bf16 elements)
  bf16* xb     = (bf16*)d_ws;                        // 8192*1024
  bf16* wqkvt  = xb     + (size_t)8192 * 1024;       // 3072*1024 (transposed)
  bf16* wprojt = wqkvt  + (size_t)3072 * 1024;       // 1024*1024 (transposed)
  bf16* qk     = wprojt + (size_t)1024 * 1024;       // 8192*2048 (Q|K)
  bf16* vtb    = qk     + (size_t)8192 * 2048;       // 64*64*2048 (V transposed)
  bf16* aout   = vtb    + (size_t)64 * 64 * 2048;    // 8192*1024

  cvt_kernel<<<(8192 * 1024) / 256, 256, 0, stream>>>(x, xb, 8192 * 1024);
  cvt_t_kernel<1024, 3072><<<(3072 * 1024) / 256, 256, 0, stream>>>(W_qkv, wqkvt);
  cvt_t_kernel<1024, 1024><<<(1024 * 1024) / 256, 256, 0, stream>>>(W_proj, wprojt);

  qkv_gemm_kernel<<<dim3(12, 256), 128, 0, stream>>>(xb, wqkvt, b_qkv, qk, vtb);
  attn_kernel<<<dim3(64, 16), 128, 0, stream>>>(qk, vtb, aout);
  proj_gemm_kernel<<<dim3(4, 256), 128, 0, stream>>>(aout, wprojt, b_proj, out);
}
